// NetGIN_12652973654600
// MI455X (gfx1250) — matrix operand order repaired and checked by reading, not compiled, not run on hardware
//
#include <hip/hip_runtime.h>

typedef __attribute__((ext_vector_type(16))) __bf16 v16bf;
typedef __attribute__((ext_vector_type(8)))  __bf16 v8bf;
typedef __attribute__((ext_vector_type(8)))  float  v8f;
typedef unsigned short u16b;

#define GG 512   // graphs

// ---------- bf16 helpers (round-to-nearest-even) ----------
__device__ __forceinline__ unsigned short f2bfu(float f) {
  unsigned u = __float_as_uint(f);
  u += 0x7FFFu + ((u >> 16) & 1u);
  return (unsigned short)(u >> 16);
}
__device__ __forceinline__ __bf16 f2bf(float f) {
  unsigned short s = f2bfu(f);
  return __builtin_bit_cast(__bf16, s);
}

// ---------- generic fill ----------
__global__ void k_fillf(float* __restrict__ p, float v, long long n) {
  long long i = (long long)blockIdx.x * blockDim.x + threadIdx.x;
  if (i < n) p[i] = v;
}

// ---------- weight conversion into WMMA-fragment-ready bf16 layout ----------
// dst element idx: i = idx&15 (elem in lane), L = (idx>>4)&31 (lane),
// t = (idx>>9)&3 (ntile), ks = idx>>11 (k-step of 32)
// maps to W[k][c] with k = ks*32 + 16*(L>>4) + i, c = t*16 + (L&15)
struct WDesc { const float* src; int kreal; int kpad; unsigned dst; };
struct WTable { WDesc d[60]; };

__global__ void k_wconv(WTable tab, unsigned short* __restrict__ pool) {
  WDesc d = tab.d[blockIdx.x];
  int total = d.kpad * 64;
  for (int idx = threadIdx.x; idx < total; idx += blockDim.x) {
    int i = idx & 15;
    int L = (idx >> 4) & 31;
    int t = (idx >> 9) & 3;
    int ks = idx >> 11;
    int k = ks * 32 + ((L >> 4) << 4) + i;
    int c = t * 16 + (L & 15);
    float v = (k < d.kreal) ? d.src[k * 64 + c] : 0.f;
    pool[d.dst + idx] = f2bfu(v);
  }
}

// ---------- agg = (1+eps)*h  (padded cols/rows zeroed) ----------
__global__ void k_init_agg(float* __restrict__ agg, const float* __restrict__ h,
                           const float* __restrict__ epsp, int Nreal, int din,
                           int hstride, int kpad, long long total) {
  long long idx = (long long)blockIdx.x * blockDim.x + threadIdx.x;
  if (idx >= total) return;
  int k = (int)(idx % kpad);
  long long n = idx / kpad;
  float v = 0.f;
  if (n < Nreal && k < din) v = (1.f + *epsp) * h[n * hstride + k];
  agg[idx] = v;
}

// ---------- edge scatter: agg[dst] += h[src] ----------
__global__ void k_scatter(float* __restrict__ agg, const float* __restrict__ h,
                          const int* __restrict__ ei, int E, int din,
                          int hstride, int kpad, int nch) {
  long long idx = (long long)blockIdx.x * blockDim.x + threadIdx.x;
  int e = (int)(idx / nch);
  if (e >= E) return;
  int c = (int)(idx % nch) * 4;
  int src = ei[e];
  int dst = ei[E + e];
  const float* hs = h + (long long)src * hstride;
  float* ad = agg + (long long)dst * kpad;
#pragma unroll
  for (int f = 0; f < 4; ++f) {
    int ff = c + f;
    if (ff < din) atomicAdd(&ad[ff], hs[ff]);
  }
}

// ---------- WMMA GEMM: C[M x 64] = A[M x K] * Wfrag + bias, fused BN column stats ----
// Block: 256 threads = 8 waves, wave -> 16 rows x 64 cols (4 WMMA n-tiles).
// Weight fragments staged to LDS once per block via async global->LDS copies.
template <bool AF32, int KSTEPS>
__global__ void __launch_bounds__(256)
k_gemm(const void* __restrict__ Aptr, int lda,
       const unsigned short* __restrict__ wfrag, const float* __restrict__ bias,
       float* __restrict__ C, float* __restrict__ colsum, float* __restrict__ colsq,
       int M) {
  __shared__ __align__(32) u16b wlds[KSTEPS * 2048];
  __shared__ float ssum[64];
  __shared__ float ssq[64];
  int tid = threadIdx.x;
  if (tid < 64) { ssum[tid] = 0.f; ssq[tid] = 0.f; }

  // --- stage all weight fragments to LDS: 16B per lane per pass (ASYNCcnt path) ---
  {
    const unsigned long long gbase = (unsigned long long)(const void*)wfrag;
#pragma unroll
    for (int it = 0; it < KSTEPS; ++it) {
      int ch = it * 256 + tid;  // 16-byte chunk id, KSTEPS*256 chunks total
      unsigned lds_off = (unsigned)(unsigned long long)(const void*)&wlds[ch * 8];
      unsigned long long ga = gbase + (unsigned long long)ch * 16ull;
      asm volatile("global_load_async_to_lds_b128 %0, %1, off"
                   :: "v"(lds_off), "v"(ga)
                   : "memory");
    }
    asm volatile("s_wait_asynccnt 0x0" ::: "memory");
  }
  __syncthreads();

  int wave = tid >> 5, lane = tid & 31;
  int half = lane >> 4, ml = lane & 15;
  long long row0 = (long long)(blockIdx.x * 8 + wave) * 16;
  long long m = row0 + ml;

  v8f acc[4];
#pragma unroll
  for (int t = 0; t < 4; ++t)
#pragma unroll
    for (int r = 0; r < 8; ++r) acc[t][r] = 0.f;

#pragma unroll
  for (int ks = 0; ks < KSTEPS; ++ks) {
    int kb = ks * 32;
    v16bf a;
    if (AF32) {
      const float* p0 = (const float*)Aptr + m * lda + kb + half * 16;
      if (ks + 1 < KSTEPS) __builtin_prefetch(p0 + 32, 0, 1);
      v8f lo = *(const v8f*)p0;
      v8f hi = *(const v8f*)(p0 + 8);
#pragma unroll
      for (int i = 0; i < 8; ++i) { a[i] = f2bf(lo[i]); a[8 + i] = f2bf(hi[i]); }
    } else {
      const u16b* p0 = (const u16b*)Aptr + m * lda + kb + half * 16;
      if (ks + 1 < KSTEPS) __builtin_prefetch(p0 + 32, 0, 1);
      v8bf lo = *(const v8bf*)p0;
      v8bf hi = *(const v8bf*)(p0 + 8);
#pragma unroll
      for (int i = 0; i < 8; ++i) { a[i] = lo[i]; a[8 + i] = hi[i]; }
    }
    const u16b* wb = wlds + ks * 2048 + lane * 16;
#pragma unroll
    for (int t = 0; t < 4; ++t) {
      v16bf b = *(const v16bf*)(wb + t * 512);
      acc[t] = __builtin_amdgcn_wmma_f32_16x16x32_bf16(
          false, a, false, b, (short)0, acc[t], false, false);
    }
  }

  // epilogue: bias add, store, column sum/sumsq (only real rows into stats)
#pragma unroll
  for (int t = 0; t < 4; ++t) {
    int col = t * 16 + ml;
    float bia = bias[col];
    float s = 0.f, q = 0.f;
#pragma unroll
    for (int r = 0; r < 8; ++r) {
      long long mr = row0 + half * 8 + r;
      float v = acc[t][r] + bia;
      C[mr * 64 + col] = v;
      if (mr < M) { s += v; q += v * v; }
    }
    atomicAdd(&ssum[col], s);
    atomicAdd(&ssq[col], q);
  }
  __syncthreads();
  if (tid < 64) {
    atomicAdd(&colsum[tid], ssum[tid]);
    atomicAdd(&colsq[tid], ssq[tid]);
  }
}

// ---------- BN finalize: scale/shift per column; re-zero accumulators ----------
__global__ void k_bnfin(float* __restrict__ colsum, float* __restrict__ colsq,
                        const float* __restrict__ g, const float* __restrict__ be,
                        float invM, float* __restrict__ sc, float* __restrict__ sh) {
  int j = threadIdx.x;
  float s = colsum[j], q = colsq[j];
  float mean = s * invM;
  float var = fmaxf(q * invM - mean * mean, 0.f);
  float scale = g[j] * rsqrtf(var + 1e-5f);
  sc[j] = scale;
  sh[j] = be[j] - mean * scale;
  colsum[j] = 0.f;
  colsq[j] = 0.f;
}

// ---------- normalize + ReLU; write bf16 (strided slice) and/or fp32 ----------
__global__ void k_norm(const float* __restrict__ C, const float* __restrict__ sc,
                       const float* __restrict__ sh, unsigned short* obf,
                       int ostride, int ooff, float* of32, long long total) {
  long long idx = (long long)blockIdx.x * blockDim.x + threadIdx.x;
  if (idx >= total) return;
  int c = (int)(idx & 63);
  long long mrow = idx >> 6;
  float v = fmaxf(0.f, C[idx] * sc[c] + sh[c]);
  if (obf) obf[mrow * (long long)ostride + ooff + c] = f2bfu(v);
  if (of32) of32[idx] = v;
}

// ---------- Set2Set ----------
__global__ void k_gates(const float* __restrict__ hh, const float* __restrict__ r,
                        const float* __restrict__ Wih, const float* __restrict__ Whh,
                        const float* __restrict__ bih, const float* __restrict__ bhh,
                        float* __restrict__ gates) {
  int t = blockIdx.x * blockDim.x + threadIdx.x;
  if (t >= GG * 256) return;
  int g = t >> 8, j = t & 255;
  const float* hg = hh + g * 64;
  const float* rg = r + g * 64;
  const float* wi = Wih + j * 128;
  const float* wh = Whh + j * 64;
  float acc = bih[j] + bhh[j];
#pragma unroll 4
  for (int k = 0; k < 64; ++k) acc += hg[k] * wi[k];
#pragma unroll 4
  for (int k = 0; k < 64; ++k) acc += rg[k] * wi[64 + k];
#pragma unroll 4
  for (int k = 0; k < 64; ++k) acc += hg[k] * wh[k];
  gates[t] = acc;
}

__global__ void k_lstm(const float* __restrict__ gates, float* __restrict__ cell,
                       float* __restrict__ hh) {
  int t = blockIdx.x * blockDim.x + threadIdx.x;
  if (t >= GG * 64) return;
  int g = t >> 6, d = t & 63;
  const float* gr = gates + g * 256;
  float gi = gr[d], gf = gr[64 + d], gc = gr[128 + d], go = gr[192 + d];
  float si = 1.f / (1.f + __expf(-gi));
  float sf = 1.f / (1.f + __expf(-gf));
  float so = 1.f / (1.f + __expf(-go));
  float c = sf * cell[t] + si * tanhf(gc);
  cell[t] = c;
  hh[t] = so * tanhf(c);
}

__device__ __forceinline__ void atomicMaxF(float* addr, float val) {
  int old = __float_as_int(*addr);
  while (__int_as_float(old) < val) {
    int prev = atomicCAS((int*)addr, old, __float_as_int(val));
    if (prev == old) break;
    old = prev;
  }
}

__global__ void k_dotmax(const float* __restrict__ h, const float* __restrict__ hh,
                         const int* __restrict__ batch, float* __restrict__ ev,
                         float* __restrict__ mmax, int N) {
  int n = blockIdx.x * blockDim.x + threadIdx.x;
  if (n >= N) return;
  int b = batch[n];
  const float* hn = h + (long long)n * 64;
  const float* hb = hh + b * 64;
  float s = 0.f;
#pragma unroll 4
  for (int k = 0; k < 64; ++k) s += hn[k] * hb[k];
  ev[n] = s;
  atomicMaxF(&mmax[b], s);
}

__global__ void k_softw(float* __restrict__ ev, const float* __restrict__ mmax,
                        const int* __restrict__ batch, float* __restrict__ denom, int N) {
  int n = blockIdx.x * blockDim.x + threadIdx.x;
  if (n >= N) return;
  int b = batch[n];
  float mv = mmax[b];
  if (!(mv > -3.0e38f)) mv = 0.f;  // "-inf" sentinel -> 0 (isfinite replacement)
  float w = __expf(ev[n] - mv);
  ev[n] = w;
  atomicAdd(&denom[b], w);
}

__global__ void k_raccum(const float* __restrict__ ev, const float* __restrict__ denom,
                         const int* __restrict__ batch, const float* __restrict__ h,
                         float* __restrict__ r, long long total) {
  long long idx = (long long)blockIdx.x * blockDim.x + threadIdx.x;
  if (idx >= total) return;
  int c = (int)(idx & 63);
  long long n = idx >> 6;
  int b = batch[n];
  float a = ev[n] / fmaxf(denom[b], 1e-16f);
  atomicAdd(&r[b * 64 + c], a * h[idx]);
}

__global__ void k_fc1(const float* __restrict__ hh, const float* __restrict__ r,
                      const float* __restrict__ W, const float* __restrict__ b,
                      float* __restrict__ tmp) {
  int t = blockIdx.x * blockDim.x + threadIdx.x;
  if (t >= GG * 64) return;
  int g = t >> 6, j = t & 63;
  const float* hg = hh + g * 64;
  const float* rg = r + g * 64;
  float acc = b[j];
#pragma unroll 4
  for (int k = 0; k < 64; ++k) acc += hg[k] * W[k * 64 + j];
#pragma unroll 4
  for (int k = 0; k < 64; ++k) acc += rg[k] * W[(64 + k) * 64 + j];
  tmp[t] = fmaxf(acc, 0.f);
}

__global__ void k_fc4(const float* __restrict__ tmp, const float* __restrict__ W,
                      const float* __restrict__ b, float* __restrict__ out) {
  int t = blockIdx.x * blockDim.x + threadIdx.x;
  if (t >= GG * 12) return;
  int g = t / 12, o = t - g * 12;
  const float* tg = tmp + g * 64;
  float acc = b[o];
#pragma unroll 4
  for (int j = 0; j < 64; ++j) acc += tg[j] * W[j * 12 + o];
  out[t] = acc;
}

// =========================================================================
extern "C" void kernel_launch(void* const* d_in, const int* in_sizes, int n_in,
                              void* d_out, int out_size, void* d_ws, size_t ws_size,
                              hipStream_t stream) {
  const int FIN = 83;
  const float* x = (const float*)d_in[0];
  const int* edges[4] = {(const int*)d_in[1], (const int*)d_in[2],
                         (const int*)d_in[3], (const int*)d_in[4]};
  const int* batch = (const int*)d_in[5];
  const int N = in_sizes[0] / FIN;
  const int E = in_sizes[1] / 2;
  const long long Mpad = ((long long)(N + 127) / 128) * 128;
  const int gblocks = (int)(Mpad / 128);

  auto F = [&](int i) { return (const float*)d_in[i]; };
  // params flattened (jax sorted-key order):
  // 6 fc1.W, 7 fc1.b, 8 fc4.W, 9 fc4.b
  // layer l base = 10+44l; conv c = +9c: [eps,W1,W2,b1,b2,be1,be2,g1,g2]; lmlp = +36: [W1,W2,b1,b2,be1,be2,g1,g2]
  // 274 W_hh, 275 W_ih, 276 b_hh, 277 b_ih
  const int p_s2s = 10 + 44 * 6;
  const float* Whh_s = F(p_s2s + 0);
  const float* Wih_s = F(p_s2s + 1);
  const float* bhh_s = F(p_s2s + 2);
  const float* bih_s = F(p_s2s + 3);

  // ---- workspace carve-up ----
  size_t off = 0;
  auto take = [&](size_t bytes) -> char* {
    off = (off + 255) & ~(size_t)255;
    char* p = (char*)d_ws + off;
    off += bytes;
    return p;
  };
  float* hbuf  = (float*)take((size_t)Mpad * 64 * 4);   // layer output h (fp32)
  float* aggb  = (float*)take((size_t)Mpad * 96 * 4);   // (1+eps)h + scatter
  float* gemC  = (float*)take((size_t)Mpad * 64 * 4);   // GEMM output
  u16b*  h1bf  = (u16b*)take((size_t)Mpad * 64 * 2);    // bf16 intermediate
  u16b*  catb  = (u16b*)take((size_t)Mpad * 256 * 2);   // bf16 concat [x1,x3,x2,x4]
  float* stats = (float*)take(256 * 4);
  float* colsum = stats, *colsq = stats + 64, *bnsc = stats + 128, *bnsh = stats + 192;
  float* ev    = (float*)take((size_t)N * 4);
  float* hhb   = (float*)take(GG * 64 * 4);
  float* cellb = (float*)take(GG * 64 * 4);
  float* rb    = (float*)take(GG * 64 * 4);
  float* gatesb= (float*)take(GG * 256 * 4);
  float* denomb= (float*)take(GG * 4);
  float* mmaxb = (float*)take(GG * 4);
  float* tmp1  = (float*)take(GG * 64 * 4);

  // ---- weight fragment pool + conversion table ----
  WTable tab;
  unsigned wpos = 0;
  unsigned w1off[6][4], w2off[6][4], wl1off[6], wl2off[6];
  int nd = 0;
  for (int l = 0; l < 6; ++l) {
    int base = 10 + 44 * l;
    int kreal = (l == 0) ? 83 : 64;
    int kp = (l == 0) ? 96 : 64;
    for (int c = 0; c < 4; ++c) {
      int cb = base + 9 * c;
      tab.d[nd++] = {F(cb + 1), kreal, kp, wpos}; w1off[l][c] = wpos; wpos += kp * 64;
      tab.d[nd++] = {F(cb + 2), 64, 64, wpos};    w2off[l][c] = wpos; wpos += 64 * 64;
    }
    int mb = base + 36;
    tab.d[nd++] = {F(mb + 0), 256, 256, wpos}; wl1off[l] = wpos; wpos += 256 * 64;
    tab.d[nd++] = {F(mb + 1), 64, 64, wpos};   wl2off[l] = wpos; wpos += 64 * 64;
  }
  u16b* wpool = (u16b*)take((size_t)wpos * 2);

  k_wconv<<<60, 256, 0, stream>>>(tab, wpool);
  k_fillf<<<1, 128, 0, stream>>>(stats, 0.f, 128);  // colsum/colsq

  const int concat_off[4] = {0, 128, 64, 192};  // concat([x1,x3,x2,x4])
  const long long ntot = Mpad * 64;
  const unsigned nblk = (unsigned)((ntot + 255) / 256);
  const float invM = 1.f / (float)N;

  const float* hin = x;
  int din = 83, hstride = 83, kp = 96;
  for (int l = 0; l < 6; ++l) {
    int base = 10 + 44 * l;
    for (int c = 0; c < 4; ++c) {
      int cb = base + 9 * c;
      long long atot = Mpad * kp;
      k_init_agg<<<(unsigned)((atot + 255) / 256), 256, 0, stream>>>(
          aggb, hin, F(cb), N, din, hstride, kp, atot);
      int nch = (din + 3) / 4;
      long long stot = (long long)E * nch;
      k_scatter<<<(unsigned)((stot + 255) / 256), 256, 0, stream>>>(
          aggb, hin, edges[c], E, din, hstride, kp, nch);
      if (kp == 96)
        k_gemm<true, 3><<<gblocks, 256, 0, stream>>>(aggb, kp, wpool + w1off[l][c],
                                                     F(cb + 3), gemC, colsum, colsq, N);
      else
        k_gemm<true, 2><<<gblocks, 256, 0, stream>>>(aggb, kp, wpool + w1off[l][c],
                                                     F(cb + 3), gemC, colsum, colsq, N);
      k_bnfin<<<1, 64, 0, stream>>>(colsum, colsq, F(cb + 7), F(cb + 5), invM, bnsc, bnsh);
      k_norm<<<nblk, 256, 0, stream>>>(gemC, bnsc, bnsh, h1bf, 64, 0, nullptr, ntot);
      k_gemm<false, 2><<<gblocks, 256, 0, stream>>>(h1bf, 64, wpool + w2off[l][c],
                                                    F(cb + 4), gemC, colsum, colsq, N);
      k_bnfin<<<1, 64, 0, stream>>>(colsum, colsq, F(cb + 8), F(cb + 6), invM, bnsc, bnsh);
      k_norm<<<nblk, 256, 0, stream>>>(gemC, bnsc, bnsh, catb, 256, concat_off[c], nullptr, ntot);
    }
    int mb = base + 36;
    k_gemm<false, 8><<<gblocks, 256, 0, stream>>>(catb, 256, wpool + wl1off[l],
                                                  F(mb + 2), gemC, colsum, colsq, N);
    k_bnfin<<<1, 64, 0, stream>>>(colsum, colsq, F(mb + 6), F(mb + 4), invM, bnsc, bnsh);
    k_norm<<<nblk, 256, 0, stream>>>(gemC, bnsc, bnsh, h1bf, 64, 0, nullptr, ntot);
    k_gemm<false, 2><<<gblocks, 256, 0, stream>>>(h1bf, 64, wpool + wl2off[l],
                                                  F(mb + 3), gemC, colsum, colsq, N);
    k_bnfin<<<1, 64, 0, stream>>>(colsum, colsq, F(mb + 7), F(mb + 5), invM, bnsc, bnsh);
    k_norm<<<nblk, 256, 0, stream>>>(gemC, bnsc, bnsh, nullptr, 0, 0, hbuf, ntot);
    hin = hbuf; din = 64; hstride = 64; kp = 64;
  }

  // ---- Set2Set readout ----
  k_fillf<<<(GG * 64 + 255) / 256, 256, 0, stream>>>(hhb, 0.f, GG * 64);
  k_fillf<<<(GG * 64 + 255) / 256, 256, 0, stream>>>(cellb, 0.f, GG * 64);
  k_fillf<<<(GG * 64 + 255) / 256, 256, 0, stream>>>(rb, 0.f, GG * 64);
  for (int step = 0; step < 6; ++step) {
    k_gates<<<(GG * 256 + 255) / 256, 256, 0, stream>>>(hhb, rb, Wih_s, Whh_s, bih_s, bhh_s, gatesb);
    k_lstm<<<(GG * 64 + 255) / 256, 256, 0, stream>>>(gatesb, cellb, hhb);
    k_fillf<<<(GG * 64 + 255) / 256, 256, 0, stream>>>(rb, 0.f, GG * 64);
    k_fillf<<<(GG + 255) / 256, 256, 0, stream>>>(denomb, 0.f, GG);
    k_fillf<<<(GG + 255) / 256, 256, 0, stream>>>(mmaxb, -3.4e38f, GG);
    k_dotmax<<<(N + 255) / 256, 256, 0, stream>>>(hbuf, hhb, batch, ev, mmaxb, N);
    k_softw<<<(N + 255) / 256, 256, 0, stream>>>(ev, mmaxb, batch, denomb, N);
    long long rtot = (long long)N * 64;
    k_raccum<<<(unsigned)((rtot + 255) / 256), 256, 0, stream>>>(ev, denomb, batch, hbuf, rb, rtot);
  }
  k_fc1<<<(GG * 64 + 255) / 256, 256, 0, stream>>>(hhb, rb, F(6), F(7), tmp1);
  k_fc4<<<(GG * 12 + 255) / 256, 256, 0, stream>>>(tmp1, F(8), F(9), (float*)d_out);

  (void)n_in; (void)out_size; (void)ws_size;
}
